// RGCN_5471788335519
// MI455X (gfx1250) — matrix operand order repaired and checked by reading, not compiled, not run on hardware
//
#include <hip/hip_runtime.h>
#include <cstdint>
#include <cstddef>

// ---------------------------------------------------------------------------
// RGCN (basis-decomposed) on MI455X / gfx1250.
// bf16 WMMA (v_wmma_f32_16x16x32_bf16) with fp32 accumulation everywhere.
// ---------------------------------------------------------------------------

typedef __attribute__((ext_vector_type(16))) __bf16       v16bf;
typedef __attribute__((ext_vector_type(8)))  float        v8f;
typedef __attribute__((ext_vector_type(4)))  unsigned int v4u;

#define RG_N_NODES 200000
#define RG_H_DIM   64
#define RG_OUT_DIM 32
#define RG_RELS    8
#define RG_BASES   8
#define RG_EDGES   1000000

// round-to-nearest-even f32 -> bf16 (data has no NaN/Inf)
__device__ __forceinline__ unsigned short rg_f2bf(float f) {
  unsigned int u = __float_as_uint(f);
  u += 0x7fffu + ((u >> 16) & 1u);
  return (unsigned short)(u >> 16);
}
__device__ __forceinline__ float rg_bf2f(unsigned short b) {
  return __uint_as_float(((unsigned int)b) << 16);
}

// ---- fold comp (x) basis -> per-relation weight matrices, stored bf16 ------
__global__ void rg_weights(const float* __restrict__ basis1, const float* __restrict__ comp1,
                           const float* __restrict__ basis2, const float* __restrict__ comp2,
                           unsigned short* __restrict__ W1b, unsigned short* __restrict__ W2b) {
  int t = blockIdx.x * blockDim.x + threadIdx.x;
  if (t < RG_RELS * RG_H_DIM * RG_H_DIM) {               // 8*64*64 = 32768
    int r = t >> 12, idx = t & 4095;
    float s = 0.f;
#pragma unroll
    for (int b = 0; b < RG_BASES; ++b) s += comp1[r * RG_BASES + b] * basis1[b * 4096 + idx];
    W1b[t] = rg_f2bf(s);
  }
  int t2 = t - RG_RELS * RG_H_DIM * RG_H_DIM;
  if (t2 >= 0 && t2 < RG_RELS * RG_H_DIM * RG_OUT_DIM) { // 8*64*32 = 16384
    int r = t2 >> 11, idx = t2 & 2047;
    float s = 0.f;
#pragma unroll
    for (int b = 0; b < RG_BASES; ++b) s += comp2[r * RG_BASES + b] * basis2[b * 2048 + idx];
    W2b[t2] = rg_f2bf(s);
  }
}

// ---- elementwise helpers ---------------------------------------------------
__global__ void rg_cast(const float* __restrict__ in, unsigned short* __restrict__ out, int n) {
  int i = blockIdx.x * blockDim.x + threadIdx.x;
  if (i < n) out[i] = rg_f2bf(in[i]);
}
__global__ void rg_zero(float* __restrict__ p, int n) {
  int i = blockIdx.x * blockDim.x + threadIdx.x;
  if (i < n) p[i] = 0.f;
}
// out_bf16[i] = bf16(in[i] + bias[i & mask])
__global__ void rg_cast_bias(const float* __restrict__ in, const float* __restrict__ bias,
                             unsigned short* __restrict__ out, int n, int mask) {
  int i = blockIdx.x * blockDim.x + threadIdx.x;
  if (i < n) out[i] = rg_f2bf(in[i] + bias[i & mask]);
}
// out[i] = bias[i & mask]   (initialize output accumulator with bias)
__global__ void rg_init_bias(float* __restrict__ out, const float* __restrict__ bias,
                             int n, int mask) {
  int i = blockIdx.x * blockDim.x + threadIdx.x;
  if (i < n) out[i] = bias[i & mask];
}

// ---- WMMA GEMM:  Y_r[M x N](bf16) = X[M x K](bf16) * W_r[K x N](bf16) ------
// Block = 128 threads = 4 waves; each wave computes 16 rows x N cols:
// A tile is loaded ONCE per k-step and reused across all N/16 WMMAs.
// grid.x = M/64, grid.y = #relations handled (W/Y advance by a full plane).
// A operand (16x32 bf16): lane m<16 holds row m, K {k0+0..7, k0+16..23};
//                         lane m>=16 holds row m-16, K {k0+8..15, k0+24..31}.
// B operand (32x16 bf16): lane l holds column N=l&15, same K order as A.
template <int K, int N>
__global__ void rg_gemm(const unsigned short* __restrict__ X,
                        const unsigned short* __restrict__ Wbase,
                        unsigned short* __restrict__ Ybase, int M) {
  constexpr int NT = N / 16;
  const int lane = threadIdx.x & 31;
  const int wave = threadIdx.x >> 5;
  const int m0   = (blockIdx.x * 4 + wave) * 16;
  const int half = lane >> 4;            // 0: lanes 0-15, 1: lanes 16-31
  const int mrow = m0 + (lane & 15);

  const unsigned short* W = Wbase + (size_t)blockIdx.y * K * N;
  unsigned short*       Y = Ybase + (size_t)blockIdx.y * M * N;

  const v8f zero = {0.f, 0.f, 0.f, 0.f, 0.f, 0.f, 0.f, 0.f};
  v8f c[NT];
#pragma unroll
  for (int t = 0; t < NT; ++t) c[t] = zero;

#pragma unroll
  for (int k0 = 0; k0 < K; k0 += 32) {
    const unsigned short* ap = X + (size_t)mrow * K + (k0 + half * 8);
    union { v4u q[2]; v16bf v; } au;
    au.q[0] = *(const v4u*)(ap);         // K = k0 + {0..7}   or {8..15}
    au.q[1] = *(const v4u*)(ap + 16);    // K = k0 + {16..23} or {24..31}
    const unsigned short* bp = W + (size_t)k0 * N + (lane & 15);
#pragma unroll
    for (int t = 0; t < NT; ++t) {
      union { unsigned short u[16]; v16bf v; } bu;
#pragma unroll
      for (int s = 0; s < 16; ++s) {
        const int kk = (s < 8) ? (half * 8 + s) : (16 + half * 8 + (s - 8));
        bu.u[s] = bp[(size_t)kk * N + t * 16];
      }
      const v16bf b = bu.v;
      c[t] = __builtin_amdgcn_wmma_f32_16x16x32_bf16(false, au.v, false, b,
                                                     (short)0, c[t], false, false);
    }
  }

  // C layout: VGPR j -> (M = j + 8*half, N = lane&15)
  const int nl = lane & 15;
#pragma unroll
  for (int t = 0; t < NT; ++t) {
#pragma unroll
    for (int j = 0; j < 8; ++j) {
      const int row = m0 + j + half * 8;
      Y[(size_t)row * N + t * 16 + nl] = rg_f2bf(c[t][j]);
    }
  }
}

// ---- edge gather/scale/scatter ---------------------------------------------
// r >= 0 : relation-filtered pass reading a single y plane (looped mode).
// r <  0 : single pass over all edges; plane selected by etype (full mode).
// Layer 1 (dim 64): one wave per edge iteration; lane handles 2 columns.
__global__ void rg_edge64(const int* __restrict__ src, const int* __restrict__ dst,
                          const int* __restrict__ et, const float* __restrict__ norm,
                          const unsigned short* __restrict__ y, size_t plane,
                          float* __restrict__ h, int nE, int r) {
  const int lane = threadIdx.x & 31;
  const int w    = (blockIdx.x * blockDim.x + threadIdx.x) >> 5;
  const int nW   = (gridDim.x * blockDim.x) >> 5;
  for (int e = w; e < nE; e += nW) {
    const int t = et[e];
    if (r >= 0 && t != r) continue;
    const int   s  = src[e];
    const int   d  = dst[e];
    const float nm = norm[e];
    const unsigned short* yp = y + (r >= 0 ? (size_t)0 : (size_t)t * plane);
    const unsigned int pk = *(const unsigned int*)(yp + ((size_t)s * 64 + lane * 2));
    const float lo = __uint_as_float(pk << 16);
    const float hi = __uint_as_float(pk & 0xffff0000u);
    atomicAdd(h + (size_t)d * 64 + lane * 2,     lo * nm);
    atomicAdd(h + (size_t)d * 64 + lane * 2 + 1, hi * nm);
  }
}
// Layer 2 (dim 32): lane handles 1 column.
__global__ void rg_edge32(const int* __restrict__ src, const int* __restrict__ dst,
                          const int* __restrict__ et, const float* __restrict__ norm,
                          const unsigned short* __restrict__ y, size_t plane,
                          float* __restrict__ o, int nE, int r) {
  const int lane = threadIdx.x & 31;
  const int w    = (blockIdx.x * blockDim.x + threadIdx.x) >> 5;
  const int nW   = (gridDim.x * blockDim.x) >> 5;
  for (int e = w; e < nE; e += nW) {
    const int t = et[e];
    if (r >= 0 && t != r) continue;
    const int   s  = src[e];
    const int   d  = dst[e];
    const float nm = norm[e];
    const unsigned short* yp = y + (r >= 0 ? (size_t)0 : (size_t)t * plane);
    const float v = rg_bf2f(yp[(size_t)s * 32 + lane]);
    atomicAdd(o + (size_t)d * 32 + lane, v * nm);
  }
}

// ---------------------------------------------------------------------------
extern "C" void kernel_launch(void* const* d_in, const int* in_sizes, int n_in,
                              void* d_out, int out_size, void* d_ws, size_t ws_size,
                              hipStream_t stream) {
  const float* emb    = (const float*)d_in[0];
  const float* basis1 = (const float*)d_in[1];
  const float* comp1  = (const float*)d_in[2];
  const float* bias1  = (const float*)d_in[3];
  const float* basis2 = (const float*)d_in[4];
  const float* comp2  = (const float*)d_in[5];
  const float* bias2  = (const float*)d_in[6];
  const int*   src    = (const int*)d_in[7];
  const int*   dst    = (const int*)d_in[8];
  const int*   etype  = (const int*)d_in[9];
  const float* norm   = (const float*)d_in[10];
  float* out = (float*)d_out;

  // workspace layout (all offsets 16B-aligned)
  char* ws = (char*)d_ws;
  unsigned short* W1b = (unsigned short*)(ws);               //  65,536 B
  unsigned short* W2b = (unsigned short*)(ws +      65536);  //  32,768 B
  unsigned short* xb  = (unsigned short*)(ws +      98304);  //  25,600,000 B
  float*          h1  = (float*)         (ws +   25698304);  //  51,200,000 B
  unsigned short* h1b = (unsigned short*)(ws +   76898304);  //  25,600,000 B
  unsigned short* yb  = (unsigned short*)(ws +  102498304);  //  y buffer

  const size_t Y_FULL1 = (size_t)RG_RELS * RG_N_NODES * RG_H_DIM;   // bf16 elems, 204.8 MB
  const bool   full    = ws_size >= (size_t)102498304 + Y_FULL1 * 2; // ~307.3 MB needed

  const int NH  = RG_N_NODES * RG_H_DIM;    // 12,800,000
  const int NO  = RG_N_NODES * RG_OUT_DIM;  //  6,400,000
  const int TPB = 256;
  const int MB  = RG_N_NODES / 64;          //  3,125 row-blocks

  rg_weights<<<192, TPB, 0, stream>>>(basis1, comp1, basis2, comp2, W1b, W2b);
  rg_cast<<<(NH + TPB - 1) / TPB, TPB, 0, stream>>>(emb, xb, NH);
  rg_zero<<<(NH + TPB - 1) / TPB, TPB, 0, stream>>>(h1, NH);

  if (full) {
    // ---- materialize all 8 relation planes; single unfiltered edge pass ----
    rg_gemm<RG_H_DIM, RG_H_DIM><<<dim3(MB, RG_RELS), 128, 0, stream>>>(
        xb, W1b, yb, RG_N_NODES);
    rg_edge64<<<4096, TPB, 0, stream>>>(src, dst, etype, norm, yb,
                                        (size_t)RG_N_NODES * RG_H_DIM, h1, RG_EDGES, -1);
  } else {
    for (int r = 0; r < RG_RELS; ++r) {
      rg_gemm<RG_H_DIM, RG_H_DIM><<<dim3(MB, 1), 128, 0, stream>>>(
          xb, W1b + (size_t)r * RG_H_DIM * RG_H_DIM, yb, RG_N_NODES);
      rg_edge64<<<2048, TPB, 0, stream>>>(src, dst, etype, norm, yb, 0, h1, RG_EDGES, r);
    }
  }

  rg_cast_bias<<<(NH + TPB - 1) / TPB, TPB, 0, stream>>>(h1, bias1, h1b, NH, RG_H_DIM - 1);
  rg_init_bias<<<(NO + TPB - 1) / TPB, TPB, 0, stream>>>(out, bias2, NO, RG_OUT_DIM - 1);

  if (full) {
    rg_gemm<RG_H_DIM, RG_OUT_DIM><<<dim3(MB, RG_RELS), 128, 0, stream>>>(
        h1b, W2b, yb, RG_N_NODES);
    rg_edge32<<<4096, TPB, 0, stream>>>(src, dst, etype, norm, yb,
                                        (size_t)RG_N_NODES * RG_OUT_DIM, out, RG_EDGES, -1);
  } else {
    for (int r = 0; r < RG_RELS; ++r) {
      rg_gemm<RG_H_DIM, RG_OUT_DIM><<<dim3(MB, 1), 128, 0, stream>>>(
          h1b, W2b + (size_t)r * RG_H_DIM * RG_OUT_DIM, yb, RG_N_NODES);
      rg_edge32<<<2048, TPB, 0, stream>>>(src, dst, etype, norm, yb, 0, out, RG_EDGES, r);
    }
  }
}
